// MultiSimilarityLoss_2705829397244
// MI455X (gfx1250) — matrix operand order repaired and checked by reading, not compiled, not run on hardware
//
#include <hip/hip_runtime.h>
#include <hip/hip_fp16.h>

typedef __attribute__((ext_vector_type(16))) _Float16 v16h;
typedef __attribute__((ext_vector_type(4)))  _Float16 v4h;
typedef __attribute__((ext_vector_type(8)))  float    v8f;

#define NB     4096      // batch
#define ND     2048      // feature dim
#define TILE   128       // block tile (M and N)
#define KSTEP  32        // WMMA K per LDS stage
#define LDSLD  40        // halfs per LDS row (32 + 8 pad, keeps 16B alignment)
#define BUFH   (TILE * LDSLD)   // halfs per LDS buffer
#define NSTAGE (ND / KSTEP)

#define THRESH    0.5f
#define MARGIN    0.1f
#define SCALE_POS 2.0f
#define SCALE_NEG 40.0f
#define ONE_EPS   (1.0f - 1e-5f)

// ---- order-preserving float<->uint keys (for atomicMin/Max on floats) ----
__device__ __forceinline__ unsigned fkey(float f) {
  unsigned u = __float_as_uint(f);
  return (u & 0x80000000u) ? ~u : (u | 0x80000000u);
}
__device__ __forceinline__ float fdecode(unsigned k) {
  unsigned u = (k & 0x80000000u) ? (k & 0x7FFFFFFFu) : ~k;
  return __uint_as_float(u);
}

__device__ __forceinline__ v16h frag16(uint4 lo, uint4 hi) {
  union { uint4 u[2]; v16h v; } t;
  t.u[0] = lo; t.u[1] = hi;
  return t.v;
}

// Low 32 bits of a flat pointer to __shared__ == LDS byte address (ISA 10.2).
__device__ __forceinline__ unsigned lds_off(const void* p) {
  return (unsigned)(uintptr_t)p;
}

// CDNA5 async Global->LDS 16B copy per lane (tracked by ASYNCcnt).
__device__ __forceinline__ void async_cp16(unsigned lds, const void* g) {
  asm volatile("global_load_async_to_lds_b128 %0, %1, off"
               :: "v"(lds), "v"((unsigned long long)(uintptr_t)g)
               : "memory");
}
__device__ __forceinline__ void wait_async0() {
  asm volatile("s_wait_asynccnt 0x0" ::: "memory");
}

// ---------------- K0: init stats + output ----------------
__global__ void k_init(unsigned* __restrict__ pminB, unsigned* __restrict__ nmaxB,
                       float* __restrict__ psum, float* __restrict__ nsum,
                       float* __restrict__ out) {
  const int i = blockIdx.x * blockDim.x + threadIdx.x;
  if (i < NB) {
    pminB[i] = 0xFF800000u;   // fkey(+inf)
    nmaxB[i] = 0x007FFFFFu;   // fkey(-inf)
    psum[i] = 0.0f;
    nsum[i] = 0.0f;
  }
  if (i == 0) out[0] = 0.0f;
}

// ---------------- K1: L2-normalize rows, emit f16 ----------------
__global__ __launch_bounds__(256) void k_normalize(const float* __restrict__ feats,
                                                   _Float16* __restrict__ Nf) {
  __shared__ float red[256];
  const int row = blockIdx.x;
  const int t = threadIdx.x;
  const float* fr = feats + (size_t)row * ND;

  float ss = 0.0f;
  for (int j = t; j < ND / 4; j += 256) {
    const float4 v = ((const float4*)fr)[j];
    ss += v.x * v.x + v.y * v.y + v.z * v.z + v.w * v.w;
  }
  red[t] = ss;
  __syncthreads();
  for (int s = 128; s > 0; s >>= 1) {
    if (t < s) red[t] += red[t + s];
    __syncthreads();
  }
  const float scale = 1.0f / (sqrtf(red[0]) + 1e-12f);

  _Float16* nr = Nf + (size_t)row * ND;
  for (int j = t; j < ND / 4; j += 256) {
    const float4 v = ((const float4*)fr)[j];
    v4h h;
    h[0] = (_Float16)(v.x * scale);
    h[1] = (_Float16)(v.y * scale);
    h[2] = (_Float16)(v.z * scale);
    h[3] = (_Float16)(v.w * scale);
    ((v4h*)nr)[j] = h;
  }
}

// ---------------- K2: WMMA GEMM (sim = Nf Nf^T), two epilogue passes ----------------
// PASS 0: per-row pos-min / neg-max (as uint keys, atomicMin/Max)
// PASS 1: per-row masked exp sums (atomicAdd)
template <int PASS>
__global__ __launch_bounds__(256) void k_gemm(const _Float16* __restrict__ Nf,
                                              const int* __restrict__ labels,
                                              unsigned* __restrict__ pminB,
                                              unsigned* __restrict__ nmaxB,
                                              float* __restrict__ psum,
                                              float* __restrict__ nsum) {
  __shared__ __align__(16) _Float16 As[2 * BUFH];   // double-buffered A tile
  __shared__ __align__(16) _Float16 Bs[2 * BUFH];   // double-buffered B tile
  __shared__ int labR[TILE], labC[TILE];
  __shared__ float pminS[TILE], nmaxS[TILE];

  const int t = threadIdx.x;
  const int lane = t & 31;
  const int wave = t >> 5;           // 8 waves
  const int wm = wave >> 1;          // 0..3 : 32-row strip
  const int wn = wave & 1;           // 0..1 : 64-col strip
  const int rowBase = blockIdx.y * TILE;
  const int colBase = blockIdx.x * TILE;

  if (t < TILE) {
    labR[t] = labels[rowBase + t];
    labC[t] = labels[colBase + t];
    if (PASS == 1) {
      pminS[t] = fdecode(pminB[rowBase + t]);
      nmaxS[t] = fdecode(nmaxB[rowBase + t]);
    }
  }

  // per-thread async-copy slot: thread t moves rows rr and rr+64, 16B column chunk c8
  const int rr = t >> 2;             // 0..63
  const int c8 = (t & 3) * 8;        // half offset 0/8/16/24

  auto issue_stage = [&](int k, int p) {
    const _Float16* gA = Nf + (size_t)(rowBase + rr) * ND + k + c8;
    const _Float16* gB = Nf + (size_t)(colBase + rr) * ND + k + c8;
    const _Float16* lA = As + p * BUFH + rr * LDSLD + c8;
    const _Float16* lB = Bs + p * BUFH + rr * LDSLD + c8;
    async_cp16(lds_off(lA), gA);
    async_cp16(lds_off(lA + 64 * LDSLD), gA + (size_t)64 * ND);
    async_cp16(lds_off(lB), gB);
    async_cp16(lds_off(lB + 64 * LDSLD), gB + (size_t)64 * ND);
  };

  v8f acc[2][4] = {};

  const int m = lane & 15;   // row/col within 16-tile
  const int h = lane >> 4;   // lane half selects K sub-block

  issue_stage(0, 0);         // prologue: stage 0 -> buffer 0

  for (int s = 0; s < NSTAGE; ++s) {
    const int p = s & 1;
    wait_async0();           // my stage-s async writes have landed in LDS
    __syncthreads();         // everyone's stage-s data visible; stage-(s-1) reads done
    if (s + 1 < NSTAGE) issue_stage((s + 1) * KSTEP, 1 - p);  // overlaps with WMMAs below

    const _Float16* Ab = As + p * BUFH;
    const _Float16* Bb = Bs + p * BUFH;

    // A fragment: 16x32 f16 -> lane m=row, half h: halfs [8h,8h+8) and [16+8h,16+8h+8)
    v16h a[2];
#pragma unroll
    for (int mt = 0; mt < 2; ++mt) {
      const _Float16* pA = Ab + (wm * 32 + mt * 16 + m) * LDSLD;
      a[mt] = frag16(*(const uint4*)(pA + h * 8), *(const uint4*)(pA + 16 + h * 8));
    }
    // B fragment: 32x16 f16 -> lane m=col, half h: halfs [8h,8h+8) and [16+8h,16+8h+8) of that row of Nf
    v16h b[4];
#pragma unroll
    for (int nt = 0; nt < 4; ++nt) {
      const _Float16* pB = Bb + (wn * 64 + nt * 16 + m) * LDSLD;
      b[nt] = frag16(*(const uint4*)(pB + h * 8), *(const uint4*)(pB + 16 + h * 8));
    }
#pragma unroll
    for (int mt = 0; mt < 2; ++mt)
#pragma unroll
      for (int nt = 0; nt < 4; ++nt)
        acc[mt][nt] = __builtin_amdgcn_wmma_f32_16x16x32_f16(
            false, a[mt], false, b[nt], (short)0, acc[mt][nt], false, false);
  }

  // ---- epilogue: C layout -> row i = base + h*8 + r, col j = base + (lane&15) ----
  const int jlane = wn * 64 + (lane & 15);
#pragma unroll
  for (int mt = 0; mt < 2; ++mt) {
#pragma unroll
    for (int r = 0; r < 8; ++r) {
      const int ii = wm * 32 + mt * 16 + h * 8 + r;
      const int gi = rowBase + ii;
      const int li = labR[ii];
      if (PASS == 0) {
        float pmn = __int_as_float(0x7F800000);  // +inf
        float nmx = -pmn;
#pragma unroll
        for (int nt = 0; nt < 4; ++nt) {
          const float s = acc[mt][nt][r];
          const int j  = jlane + nt * 16;
          const int gj = colBase + j;
          if (li == labC[j]) {
            if (gi != gj && s < ONE_EPS) pmn = fminf(pmn, s);
          } else {
            nmx = fmaxf(nmx, s);
          }
        }
#pragma unroll
        for (int d = 1; d < 16; d <<= 1) {
          pmn = fminf(pmn, __shfl_xor(pmn, d, 32));
          nmx = fmaxf(nmx, __shfl_xor(nmx, d, 32));
        }
        if ((lane & 15) == 0) {
          atomicMin(&pminB[gi], fkey(pmn));
          atomicMax(&nmaxB[gi], fkey(nmx));
        }
      } else {
        const float rowPmin = pminS[ii];
        const float rowNmax = nmaxS[ii];
        float ps = 0.0f, ns = 0.0f;
#pragma unroll
        for (int nt = 0; nt < 4; ++nt) {
          const float s = acc[mt][nt][r];
          const int j  = jlane + nt * 16;
          const int gj = colBase + j;
          if (li == labC[j]) {
            if (gi != gj && s < ONE_EPS && (s - MARGIN) < rowNmax)
              ps += expf(-SCALE_POS * (s - THRESH));
          } else {
            if ((s + MARGIN) > rowPmin)
              ns += expf(SCALE_NEG * (s - THRESH));
          }
        }
#pragma unroll
        for (int d = 1; d < 16; d <<= 1) {
          ps += __shfl_xor(ps, d, 32);
          ns += __shfl_xor(ns, d, 32);
        }
        if ((lane & 15) == 0) {
          atomicAdd(&psum[gi], ps);
          atomicAdd(&nsum[gi], ns);
        }
      }
    }
  }
}

// ---------------- K3: finalize per-row loss, reduce to scalar ----------------
__global__ void k_finalize(const float* __restrict__ psum, const float* __restrict__ nsum,
                           float* __restrict__ out) {
  const int i = blockIdx.x * blockDim.x + threadIdx.x;
  if (i >= NB) return;
  const float ps = psum[i], ns = nsum[i];
  // exp(...) > 0 strictly, so sum > 0 <=> at least one kept pair
  float c = 0.0f;
  if (ps > 0.0f && ns > 0.0f)
    c = log1pf(ps) * (1.0f / SCALE_POS) + log1pf(ns) * (1.0f / SCALE_NEG);
  atomicAdd(out, c * (1.0f / (float)NB));
}

extern "C" void kernel_launch(void* const* d_in, const int* in_sizes, int n_in,
                              void* d_out, int out_size, void* d_ws, size_t ws_size,
                              hipStream_t stream) {
  const float* feats = (const float*)d_in[0];
  const int* labels  = (const int*)d_in[1];   // labels fit int32 (values < 256)
  float* out = (float*)d_out;

  char* ws = (char*)d_ws;
  _Float16* Nf     = (_Float16*)ws;                                // 4096*2048*2 = 16 MB
  unsigned* pminB  = (unsigned*)(ws + (size_t)NB * ND * 2);        // 16 KB
  unsigned* nmaxB  = pminB + NB;                                   // 16 KB
  float*    psum   = (float*)(nmaxB + NB);                         // 16 KB
  float*    nsum   = psum + NB;                                    // 16 KB

  k_init<<<(NB + 255) / 256, 256, 0, stream>>>(pminB, nmaxB, psum, nsum, out);
  k_normalize<<<NB, 256, 0, stream>>>(feats, Nf);

  dim3 grid(NB / TILE, NB / TILE);   // 32 x 32 blocks of 128x128
  k_gemm<0><<<grid, 256, 0, stream>>>(Nf, labels, pminB, nmaxB, psum, nsum);
  k_gemm<1><<<grid, 256, 0, stream>>>(Nf, labels, pminB, nmaxB, psum, nsum);

  k_finalize<<<(NB + 255) / 256, 256, 0, stream>>>(psum, nsum, out);
}
